// paralleltokenmixer_47150150975769
// MI455X (gfx1250) — compile-verified, matrix operand order checked by reading, not run on hardware
//
#include <hip/hip_runtime.h>
#include <hip/hip_bf16.h>

typedef __attribute__((ext_vector_type(16))) _Float16 v16h;
typedef __attribute__((ext_vector_type(8)))  _Float16 v8h;
typedef __attribute__((ext_vector_type(8)))  float    v8f;
typedef __attribute__((ext_vector_type(4)))  int      i32x4;

#define TOK 131072            // 2048 windows * 64 tokens
#define IMG 33554432          // 32*256*64*64
#define XA  16777216          // 131072*128

#if defined(__has_builtin)
#if __has_builtin(__builtin_amdgcn_global_load_async_to_lds_b128)
#define USE_ASYNC 1
#endif
#endif
#ifndef USE_ASYNC
#define USE_ASYNC 0
#endif

__device__ inline v8f wmma_f16(v16h a, v16h b, v8f c) {
  return __builtin_amdgcn_wmma_f32_16x16x32_f16(false, a, false, b, (short)0, c, false, false);
}

__device__ inline float gelu_f(float x) {
  return 0.5f * x * (1.f + erff(x * 0.70710678118654752440f));
}

#if USE_ASYNC
typedef __attribute__((address_space(1))) i32x4* gvec_t;   // global int4*
typedef __attribute__((address_space(3))) i32x4* lvec_t;   // LDS int4*
__device__ inline gvec_t as_global(const void* p) {
  return (gvec_t)(unsigned long long)p;
}
__device__ inline lvec_t as_lds(const void* p) {
  // low 32 bits of a generic shared address are the LDS byte address (ISA 10.2)
  return (lvec_t)(unsigned int)(unsigned long long)p;
}
#endif

// Fragment loads from row-major "fragment storage": 16 rows x (>=32) halves.
// Per ISA 7.12.2 wave32 layout: lane<16 -> row=lane, K 0..7 & 16..23;
// lane>=16 -> row=lane-16, K 8..15 & 24..31. ld must be a multiple of 8 halves,
// base 16B aligned -> two ds_load_b128 per fragment.
__device__ inline v16h ldfrag16(const _Float16* base, int ld, int lane) {
  int m = lane & 15;
  int kb = (lane & 16) ? 8 : 0;
  const _Float16* p = base + m * ld + kb;
  v8h lo = *(const v8h*)p;
  v8h hi = *(const v8h*)(p + 16);
  return __builtin_shufflevector(lo, hi, 0,1,2,3,4,5,6,7,8,9,10,11,12,13,14,15);
}
// Same, but K limited to 16 (upper half of the 32-K fragment zeroed).
__device__ inline v16h ldfrag8z(const _Float16* base, int ld, int lane) {
  int m = lane & 15;
  int kb = (lane & 16) ? 8 : 0;
  v8h lo = *(const v8h*)(base + m * ld + kb);
  v8h z = {};
  return __builtin_shufflevector(lo, z, 0,1,2,3,4,5,6,7,8,9,10,11,12,13,14,15);
}

// ---------------- generic WMMA GEMM: C = A[M,K] * B[K,N] + bias ----------------
// grid = (N/64, M/64, batch). bias_mode: 0 none, 1 per-col, 2 per-row.
__global__ __launch_bounds__(256) void k_gemm(
    const float* __restrict__ A, const float* __restrict__ B,
    const float* __restrict__ bias, float* __restrict__ C,
    int N, int K, int bias_mode,
    unsigned long long sAb, unsigned long long sBb, unsigned long long sCb)
{
  __shared__ alignas(16) _Float16 As[64][48];   // 64 rows x 32 K (f16), padded
  __shared__ alignas(16) _Float16 Bt[64][48];   // 64 N-rows x 32 K (f16), transposed
#if USE_ASYNC
  __shared__ alignas(16) float Araw[2048];      // 64x32 f32
  __shared__ alignas(16) float Braw[2048];      // 32x64 f32
#endif
  const int tid = threadIdx.x;
  const int lane = tid & 31;
  const int wave = tid >> 5;
  const int rt = wave & 3;     // row tile (16 rows)
  const int ch = wave >> 2;    // column half (32 cols)
  const int bm = blockIdx.y * 64;
  const int bn = blockIdx.x * 64;
  const size_t ab = (size_t)blockIdx.z * sAb;
  const size_t bb = (size_t)blockIdx.z * sBb;
  const size_t cb = (size_t)blockIdx.z * sCb;

  v8f acc0 = {}; v8f acc1 = {};
  for (int k0 = 0; k0 < K; k0 += 32) {
    if (k0 + 32 < K) {   // prefetch next K tiles (global_prefetch_b8)
      __builtin_prefetch(A + ab + (size_t)(bm + (tid & 63)) * K + (k0 + 32), 0, 0);
      __builtin_prefetch(B + bb + (size_t)(k0 + 32 + (tid & 31)) * N + bn, 0, 0);
    }
#if USE_ASYNC
    // 16B per lane async copies: A tile (8KB) + B tile (8KB), ASYNCcnt-tracked
    for (int c = tid; c < 512; c += 256) {
      const float* gA = A + ab + (size_t)(bm + (c >> 3)) * K + (k0 + ((c & 7) << 2));
      __builtin_amdgcn_global_load_async_to_lds_b128(as_global(gA), as_lds(&Araw[c << 2]), 0, 0);
      const float* gB = B + bb + (size_t)(k0 + (c >> 4)) * N + (bn + ((c & 15) << 2));
      __builtin_amdgcn_global_load_async_to_lds_b128(as_global(gB), as_lds(&Braw[c << 2]), 0, 0);
    }
    asm volatile("s_wait_asynccnt 0x0" ::: "memory");
    __syncthreads();
    {   // convert A raw f32 -> f16 rows (contiguous, b128 in / b128 out)
      int r = tid >> 2, c0 = (tid & 3) << 3;
      const float* pr = &Araw[r * 32 + c0];
      v8h h;
#pragma unroll
      for (int i = 0; i < 8; ++i) h[i] = (_Float16)pr[i];
      *(v8h*)&As[r][c0] = h;
    }
    for (int c = tid; c < 512; c += 256) {   // convert + transpose B
      int k2 = c >> 4, n0 = (c & 15) << 2;
      const float* pr = &Braw[k2 * 64 + n0];
#pragma unroll
      for (int j = 0; j < 4; ++j) Bt[n0 + j][k2] = (_Float16)pr[j];
    }
#else
    {   // direct staging: float4 global loads, v8h LDS store for A
      int r = tid >> 2, c0 = (tid & 3) << 3;
      const float* gA = A + ab + (size_t)(bm + r) * K + (k0 + c0);
      float4 f0 = *(const float4*)gA;
      float4 f1 = *(const float4*)(gA + 4);
      v8h h;
      h[0] = (_Float16)f0.x; h[1] = (_Float16)f0.y; h[2] = (_Float16)f0.z; h[3] = (_Float16)f0.w;
      h[4] = (_Float16)f1.x; h[5] = (_Float16)f1.y; h[6] = (_Float16)f1.z; h[7] = (_Float16)f1.w;
      *(v8h*)&As[r][c0] = h;
    }
    for (int c = tid; c < 512; c += 256) {   // B: coalesced float4 loads, transposed stores
      int k2 = c >> 4, n0 = (c & 15) << 2;
      float4 f = *(const float4*)(B + bb + (size_t)(k0 + k2) * N + (bn + n0));
      Bt[n0 + 0][k2] = (_Float16)f.x;
      Bt[n0 + 1][k2] = (_Float16)f.y;
      Bt[n0 + 2][k2] = (_Float16)f.z;
      Bt[n0 + 3][k2] = (_Float16)f.w;
    }
#endif
    __syncthreads();
    v16h a  = ldfrag16(&As[rt * 16][0], 48, lane);
    v16h b0 = ldfrag16(&Bt[ch * 32][0], 48, lane);
    v16h b1 = ldfrag16(&Bt[ch * 32 + 16][0], 48, lane);
    acc0 = wmma_f16(a, b0, acc0);
    acc1 = wmma_f16(a, b1, acc1);
    __syncthreads();
  }
  int col0 = bn + ch * 32 + (lane & 15);
  int rb = bm + rt * 16 + ((lane & 16) ? 8 : 0);
#pragma unroll
  for (int i = 0; i < 8; ++i) {
    int gr = rb + i;
    float v0 = acc0[i], v1 = acc1[i];
    if (bias_mode == 1) { v0 += bias[col0]; v1 += bias[col0 + 16]; }
    else if (bias_mode == 2) { v0 += bias[gr]; v1 += bias[gr]; }
    C[cb + (size_t)gr * N + col0]      = v0;
    C[cb + (size_t)gr * N + col0 + 16] = v1;
  }
}

// ---------------- LayerNorm over last dim (W = 128 or 256) ----------------
__global__ __launch_bounds__(128) void k_ln(
    const float* __restrict__ x, const float* __restrict__ g, const float* __restrict__ bb,
    float* __restrict__ dst, int W, int dstride, int doff)
{
  __shared__ float xb[256];
  __shared__ float r1[128], r2[128];
  int row = blockIdx.x, tid = threadIdx.x;
  float s1 = 0.f, s2 = 0.f;
  for (int c = tid; c < W; c += 128) {
    float v = x[(size_t)row * W + c];
    xb[c] = v; s1 += v; s2 += v * v;
  }
  r1[tid] = s1; r2[tid] = s2;
  __syncthreads();
  for (int off = 64; off > 0; off >>= 1) {
    if (tid < off) { r1[tid] += r1[tid + off]; r2[tid] += r2[tid + off]; }
    __syncthreads();
  }
  float mean = r1[0] / (float)W;
  float var = r2[0] / (float)W - mean * mean;
  float rstd = rsqrtf(var + 1e-5f);
  for (int c = tid; c < W; c += 128)
    dst[(size_t)row * dstride + doff + c] = (xb[c] - mean) * rstd * g[c] + bb[c];
}

// ---------------- BatchNorm channel statistics (biased var) ----------------
__global__ __launch_bounds__(256) void k_bnstats(
    const float* __restrict__ x, float* __restrict__ mean, float* __restrict__ rstd,
    int C, int HW, int NB)
{
  __shared__ float r1[256], r2[256];
  int c = blockIdx.x, tid = threadIdx.x;
  float s1 = 0.f, s2 = 0.f;
  int total = NB * HW;
  for (int i = tid; i < total; i += 256) {
    int b = i / HW, p = i - b * HW;
    float v = x[((size_t)b * C + c) * HW + p];
    s1 += v; s2 += v * v;
  }
  r1[tid] = s1; r2[tid] = s2;
  __syncthreads();
  for (int off = 128; off > 0; off >>= 1) {
    if (tid < off) { r1[tid] += r1[tid + off]; r2[tid] += r2[tid + off]; }
    __syncthreads();
  }
  if (tid == 0) {
    float m = r1[0] / (float)total;
    float v = r2[0] / (float)total - m * m;
    mean[c] = m; rstd[c] = rsqrtf(v + 1e-5f);
  }
}

// ---------------- window reverse: tokens (TOK,256) -> NCHW (32,256,64,64) ----------------
__global__ void k_winrev(const float* __restrict__ src, float* __restrict__ dst) {
  size_t idx = (size_t)blockIdx.x * 256 + threadIdx.x;
  int b = (int)(idx >> 20);
  int c = (int)(idx >> 12) & 255;
  int h = (int)(idx >> 6) & 63;
  int w = (int)idx & 63;
  int r = ((b * 8 + (h >> 3)) * 8 + (w >> 3)) * 64 + ((h & 7) * 8) + (w & 7);
  dst[idx] = src[(size_t)r * 256 + c];
}

// ---------------- depthwise 3x3 conv + bias ----------------
__global__ void k_dwconv(const float* __restrict__ x, const float* __restrict__ w,
                         const float* __restrict__ bias, float* __restrict__ y) {
  size_t idx = (size_t)blockIdx.x * 256 + threadIdx.x;
  int c = (int)(idx >> 12) & 255;
  int h = (int)(idx >> 6) & 63;
  int ww = (int)idx & 63;
  size_t base = idx - (size_t)(h * 64 + ww);
  float acc = bias[c];
#pragma unroll
  for (int kh = 0; kh < 3; ++kh) {
    int hh = h + kh - 1;
    if (hh < 0 || hh > 63) continue;
#pragma unroll
    for (int kw = 0; kw < 3; ++kw) {
      int w2 = ww + kw - 1;
      if (w2 < 0 || w2 > 63) continue;
      acc += x[base + hh * 64 + w2] * w[c * 9 + kh * 3 + kw];
    }
  }
  y[idx] = acc;
}

// ---------------- BN apply + exact GELU, in place, NCHW ----------------
__global__ void k_bngelu(float* __restrict__ x, const float* __restrict__ mn,
                         const float* __restrict__ rs, const float* __restrict__ g,
                         const float* __restrict__ bb, int C, int HW) {
  size_t idx = (size_t)blockIdx.x * 256 + threadIdx.x;
  int c = (int)((idx / (size_t)HW) % (size_t)C);
  float v = (x[idx] - mn[c]) * rs[c] * g[c] + bb[c];
  x[idx] = gelu_f(v);
}

// ---------------- global average pool per (b,c): 4096 -> 1 ----------------
__global__ __launch_bounds__(256) void k_pool(const float* __restrict__ x, float* __restrict__ out) {
  __shared__ float r1[256];
  int blk = blockIdx.x, tid = threadIdx.x;
  size_t base = (size_t)blk * 4096;
  float s = 0.f;
  for (int p = tid; p < 4096; p += 256) s += x[base + p];
  r1[tid] = s;
  __syncthreads();
  for (int off = 128; off > 0; off >>= 1) {
    if (tid < off) r1[tid] += r1[tid + off];
    __syncthreads();
  }
  if (tid == 0) out[blk] = r1[0] * (1.f / 4096.f);
}

// ---------------- channel interaction (SE): 256->32 BN GELU ->128 sigmoid gate ----------------
__global__ __launch_bounds__(1024) void k_ci(
    const float* __restrict__ pool, const float* __restrict__ w1, const float* __restrict__ b1,
    const float* __restrict__ bng, const float* __restrict__ bnb,
    const float* __restrict__ w2, const float* __restrict__ b2,
    float* __restrict__ gate)
{
  __shared__ float y1[1024], z[1024], mn[32], rs[32];
  int t = threadIdx.x;
  int b = t >> 5, o = t & 31;
  float acc = b1[o];
  for (int c = 0; c < 256; ++c) acc += pool[b * 256 + c] * w1[o * 256 + c];
  y1[t] = acc;
  __syncthreads();
  if (t < 32) {
    float s1 = 0.f, s2 = 0.f;
    for (int b2i = 0; b2i < 32; ++b2i) { float v = y1[b2i * 32 + t]; s1 += v; s2 += v * v; }
    float m = s1 / 32.f; float va = s2 / 32.f - m * m;
    mn[t] = m; rs[t] = rsqrtf(va + 1e-5f);
  }
  __syncthreads();
  z[t] = gelu_f((y1[t] - mn[o]) * rs[o] * bng[o] + bnb[o]);
  __syncthreads();
  for (int idx = t; idx < 4096; idx += 1024) {
    int b3 = idx >> 7, o2 = idx & 127;
    float a2 = b2[o2];
    for (int oo = 0; oo < 32; ++oo) a2 += w2[o2 * 32 + oo] * z[b3 * 32 + oo];
    gate[idx] = 1.f / (1.f + expf(-a2));
  }
}

// ---------------- relative position bias table -> (8,64,64) ----------------
__global__ void k_biasf(const float* __restrict__ rpb, float* __restrict__ bf) {
  int idx = blockIdx.x * 256 + threadIdx.x;   // < 32768
  int h = idx >> 12;
  int n = (idx >> 6) & 63, m = idx & 63;
  int di = (n >> 3) - (m >> 3) + 7;
  int dj = (n & 7) - (m & 7) + 7;
  bf[idx] = rpb[(di * 15 + dj) * 8 + h];
}

// ---------------- fused window attention per (window, head) ----------------
__global__ __launch_bounds__(128) void k_attn(
    const float* __restrict__ qkv, const float* __restrict__ gate,
    const float* __restrict__ bf, float* __restrict__ out)
{
  __shared__ alignas(16) _Float16 Qs[64][24], Ks[64][24];
  __shared__ alignas(16) _Float16 Vt[16][80];   // V transposed: [d][n]
  __shared__ float Sm[64][66];
  __shared__ alignas(16) _Float16 Ps[64][80];
  const int b_ = blockIdx.x, h = blockIdx.y;
  const int tid = threadIdx.x, lane = tid & 31, wave = tid >> 5;
  const int bi = b_ >> 6;

  for (int e = tid; e < 1024; e += 128) {
    int n = e >> 4, d = e & 15;
    size_t r = (size_t)(b_ * 64 + n) * 384;
    Qs[n][d] = (_Float16)(qkv[r + h * 16 + d] * 0.25f);
    Ks[n][d] = (_Float16)(qkv[r + 128 + h * 16 + d]);
    Vt[d][n] = (_Float16)(qkv[r + 256 + h * 16 + d] * gate[bi * 128 + h * 16 + d]);
  }
  __syncthreads();

  // S = Q K^T  (K=16 zero-padded to 32)
  {
    v16h a = ldfrag8z(&Qs[wave * 16][0], 24, lane);
#pragma unroll
    for (int ct = 0; ct < 4; ++ct) {
      v16h bq = ldfrag8z(&Ks[ct * 16][0], 24, lane);
      v8f c = {};
      c = wmma_f16(a, bq, c);
      int ro = wave * 16 + ((lane & 16) ? 8 : 0);
      int col = ct * 16 + (lane & 15);
#pragma unroll
      for (int i = 0; i < 8; ++i) Sm[ro + i][col] = c[i];
    }
  }
  __syncthreads();

  // softmax rows with relative position bias
  if (tid < 64) {
    int n = tid;
    const float* bp = bf + ((size_t)h * 64 + n) * 64;
    float mx = -3.0e38f;
    for (int m2 = 0; m2 < 64; ++m2) {
      float s = Sm[n][m2] + bp[m2];
      Sm[n][m2] = s;
      mx = fmaxf(mx, s);
    }
    float sum = 0.f;
    for (int m2 = 0; m2 < 64; ++m2) {
      float e = expf(Sm[n][m2] - mx);
      Sm[n][m2] = e; sum += e;
    }
    float inv = 1.f / sum;
    for (int m2 = 0; m2 < 64; ++m2) Ps[n][m2] = (_Float16)(Sm[n][m2] * inv);
  }
  __syncthreads();

  // O = P V  (M=64, N=16, K=64)
  {
    int d = lane & 15;
    v8f acc = {};
#pragma unroll
    for (int k0 = 0; k0 < 64; k0 += 32) {
      v16h a  = ldfrag16(&Ps[wave * 16][k0], 80, lane);
      v16h bv = ldfrag16(&Vt[0][k0], 80, lane);
      acc = wmma_f16(a, bv, acc);
    }
    int ro = wave * 16 + ((lane & 16) ? 8 : 0);
#pragma unroll
    for (int i = 0; i < 8; ++i) {
      size_t row = (size_t)b_ * 64 + ro + i;
      out[row * 128 + h * 16 + d] = acc[i];
    }
  }
}

// ---------------- spatial interaction stage 1: 1x1 conv 128->16 with window gather ----------------
__global__ void k_si1(const float* __restrict__ ao, const float* __restrict__ w,
                      const float* __restrict__ bias, float* __restrict__ y) {
  size_t idx = (size_t)blockIdx.x * 256 + threadIdx.x;  // < 32*16*4096
  int b = (int)(idx >> 16);
  int o = (int)(idx >> 12) & 15;
  int p = (int)idx & 4095;
  int h = p >> 6, ww = p & 63;
  int r = ((b * 8 + (h >> 3)) * 8 + (ww >> 3)) * 64 + ((h & 7) * 8) + (ww & 7);
  const float* xr = ao + (size_t)r * 128;
  float acc = bias[o];
  for (int c = 0; c < 128; ++c) acc += w[o * 128 + c] * xr[c];
  y[idx] = acc;
}

// ---------------- spatial interaction stage 2: BN+GELU+1x1(16->1)+sigmoid ----------------
__global__ void k_si2(const float* __restrict__ s1, const float* __restrict__ mn,
                      const float* __restrict__ rs, const float* __restrict__ g,
                      const float* __restrict__ bb, const float* __restrict__ w2,
                      const float* __restrict__ b2, float* __restrict__ gsp) {
  int idx = blockIdx.x * 256 + threadIdx.x;   // < 131072
  int b = idx >> 12, p = idx & 4095;
  float acc = b2[0];
  for (int o = 0; o < 16; ++o) {
    float v = s1[((size_t)b * 16 + o) * 4096 + p];
    v = (v - mn[o]) * rs[o] * g[o] + bb[o];
    acc += w2[o] * gelu_f(v);
  }
  gsp[idx] = 1.f / (1.f + expf(-acc));
}

// ---------------- spatial gate multiply in place on (32,128,4096) ----------------
__global__ void k_gmul(float* __restrict__ x, const float* __restrict__ gsp) {
  size_t idx = (size_t)blockIdx.x * 256 + threadIdx.x;  // 2^24
  int b = (int)(idx >> 19);
  int p = (int)idx & 4095;
  x[idx] *= gsp[(b << 12) + p];
}

// ---------------- BN apply + window partition into concat[:,0:128] ----------------
__global__ void k_bnpart(const float* __restrict__ x, const float* __restrict__ mn,
                         const float* __restrict__ rs, const float* __restrict__ g,
                         const float* __restrict__ bb, float* __restrict__ dst) {
  size_t idx = (size_t)blockIdx.x * 256 + threadIdx.x;  // 2^24
  int b = (int)(idx >> 19);
  int c = (int)(idx >> 12) & 127;
  int p = (int)idx & 4095;
  int h = p >> 6, w = p & 63;
  float v = (x[idx] - mn[c]) * rs[c] * g[c] + bb[c];
  int r = ((b * 8 + (h >> 3)) * 8 + (w >> 3)) * 64 + ((h & 7) * 8) + (w & 7);
  dst[(size_t)r * 256 + c] = v;
}

extern "C" void kernel_launch(void* const* d_in, const int* in_sizes, int n_in,
                              void* d_out, int out_size, void* d_ws, size_t ws_size,
                              hipStream_t stream) {
  const float* x           = (const float*)d_in[0];
  // d_in[1]=H, d_in[2]=W (static 64)
  const float* proj_attn_w = (const float*)d_in[3];
  const float* proj_attn_b = (const float*)d_in[4];
  const float* ln_attn_g   = (const float*)d_in[5];
  const float* ln_attn_b   = (const float*)d_in[6];
  const float* proj_cnn_w  = (const float*)d_in[7];
  const float* proj_cnn_b  = (const float*)d_in[8];
  const float* ln_cnn_g    = (const float*)d_in[9];
  const float* ln_cnn_b    = (const float*)d_in[10];
  const float* dw_w        = (const float*)d_in[11];
  const float* dw_b        = (const float*)d_in[12];
  const float* bn_dw_g     = (const float*)d_in[13];
  const float* bn_dw_b     = (const float*)d_in[14];
  const float* ci1_w       = (const float*)d_in[15];
  const float* ci1_b       = (const float*)d_in[16];
  const float* bn_ci_g     = (const float*)d_in[17];
  const float* bn_ci_b     = (const float*)d_in[18];
  const float* ci2_w       = (const float*)d_in[19];
  const float* ci2_b       = (const float*)d_in[20];
  const float* projc_w     = (const float*)d_in[21];
  const float* projc_b     = (const float*)d_in[22];
  const float* bn_conv_g   = (const float*)d_in[23];
  const float* bn_conv_b   = (const float*)d_in[24];
  const float* qkv_w       = (const float*)d_in[25];
  const float* qkv_b       = (const float*)d_in[26];
  const float* si1_w       = (const float*)d_in[27];
  const float* si1_b       = (const float*)d_in[28];
  const float* bn_si_g     = (const float*)d_in[29];
  const float* bn_si_b     = (const float*)d_in[30];
  const float* si2_w       = (const float*)d_in[31];
  const float* si2_b       = (const float*)d_in[32];
  const float* ln_out_g    = (const float*)d_in[33];
  const float* ln_out_b    = (const float*)d_in[34];
  const float* proj_w      = (const float*)d_in[35];
  const float* proj_b      = (const float*)d_in[36];
  const float* rpb_table   = (const float*)d_in[37];

  float* ws  = (float*)d_ws;
  float* xc  = ws;                           // TOK*256 (proj_cnn), later reused as qkv TOK*384
  float* qkv = ws;
  float* img = ws + (size_t)TOK * 384;       // IMG floats: xc_img, later concat buffer
  float* xdw = img + (size_t)IMG;            // IMG floats: dwconv -> BN+GELU (xg)
  float* xa  = xdw + (size_t)IMG;            // XA floats: x_atten, later x_cnn128
  float* ao  = xa + (size_t)XA;              // XA floats: attention output
  float* cip = ao + (size_t)XA;              // 8192
  float* gate = cip + 8192;                  // 4096
  float* bfp = gate + 4096;                  // 32768
  float* s1b = bfp + 32768;                  // 2097152 (si1)
  float* gsp = s1b + 2097152;                // 131072
  float* m1 = gsp + 131072; float* r1 = m1 + 256;
  float* m2 = r1 + 256;     float* r2 = m2 + 16;
  float* m3 = r2 + 16;      float* r3 = m3 + 128;

  // 1) parallel projections + LayerNorms
  k_gemm<<<dim3(4, TOK / 64, 1), 256, 0, stream>>>(x, proj_cnn_w, proj_cnn_b, xc, 256, 256, 1, 0, 0, 0);
  k_ln<<<TOK, 128, 0, stream>>>(xc, ln_cnn_g, ln_cnn_b, xc, 256, 256, 0);
  k_winrev<<<IMG / 256, 256, 0, stream>>>(xc, img);
  k_gemm<<<dim3(2, TOK / 64, 1), 256, 0, stream>>>(x, proj_attn_w, proj_attn_b, xa, 128, 256, 1, 0, 0, 0);
  k_ln<<<TOK, 128, 0, stream>>>(xa, ln_attn_g, ln_attn_b, xa, 128, 128, 0);

  // 2) conv branch: depthwise conv -> BN -> GELU
  k_dwconv<<<IMG / 256, 256, 0, stream>>>(img, dw_w, dw_b, xdw);
  k_bnstats<<<256, 256, 0, stream>>>(xdw, m1, r1, 256, 4096, 32);
  k_bngelu<<<IMG / 256, 256, 0, stream>>>(xdw, m1, r1, bn_dw_g, bn_dw_b, 256, 4096);

  // 3) channel interaction (SE) -> sigmoid gate
  k_pool<<<32 * 256, 256, 0, stream>>>(xdw, cip);
  k_ci<<<1, 1024, 0, stream>>>(cip, ci1_w, ci1_b, bn_ci_g, bn_ci_b, ci2_w, ci2_b, gate);

  // 4) attention: qkv GEMM (overwrites xc slot), bias table, fused attention
  k_gemm<<<dim3(6, TOK / 64, 1), 256, 0, stream>>>(xa, qkv_w, qkv_b, qkv, 384, 128, 1, 0, 0, 0);
  k_biasf<<<128, 256, 0, stream>>>(rpb_table, bfp);
  k_attn<<<dim3(2048, 8), 128, 0, stream>>>(qkv, gate, bfp, ao);

  // 5) projc 1x1 conv 256->128 (batched GEMM, per-row bias) into xa slot
  k_gemm<<<dim3(64, 2, 32), 256, 0, stream>>>(projc_w, xdw, projc_b, xa, 4096, 256, 2,
                                              0ULL, (unsigned long long)256 * 4096,
                                              (unsigned long long)128 * 4096);

  // 6) spatial interaction gate
  k_si1<<<2097152 / 256, 256, 0, stream>>>(ao, si1_w, si1_b, s1b);
  k_bnstats<<<16, 256, 0, stream>>>(s1b, m2, r2, 16, 4096, 32);
  k_si2<<<131072 / 256, 256, 0, stream>>>(s1b, m2, r2, bn_si_g, bn_si_b, si2_w, si2_b, gsp);
  k_gmul<<<(32 * 128 * 4096) / 256, 256, 0, stream>>>(xa, gsp);

  // 7) BN conv branch + window partition into concat[:,0:128]; LN attn into concat[:,128:256]
  k_bnstats<<<128, 256, 0, stream>>>(xa, m3, r3, 128, 4096, 32);
  k_bnpart<<<(32 * 128 * 4096) / 256, 256, 0, stream>>>(xa, m3, r3, bn_conv_g, bn_conv_b, img);
  k_ln<<<TOK, 128, 0, stream>>>(ao, ln_out_g, ln_out_b, img, 128, 256, 128);

  // 8) final projection into d_out
  k_gemm<<<dim3(4, TOK / 64, 1), 256, 0, stream>>>(img, proj_w, proj_b, (float*)d_out, 256, 256, 1, 0, 0, 0);
}
